// PoETS_37297495999163
// MI455X (gfx1250) — compile-verified, matrix operand order checked by reading, not run on hardware
//
#include <hip/hip_runtime.h>

#define L_LAYERS 6
#define Bq 8
#define Mm 2048
#define Dd 768
#define Hh 12
#define DHd 64
#define FFd 3072
#define Vv 30
#define ASPLIT 4
#define ACHUNK 513              // ceil(2049 / 4)

typedef __attribute__((ext_vector_type(16))) _Float16 v16h;
typedef __attribute__((ext_vector_type(8)))  _Float16 v8h;
typedef __attribute__((ext_vector_type(8)))  float    v8f;
typedef __attribute__((ext_vector_type(4)))  float    v4f;

union Frag16 { v16h v; v8h h8[2]; };

__device__ __forceinline__ void atomic_add_f32(float* p, float v) {
    __hip_atomic_fetch_add(p, v, __ATOMIC_RELAXED, __HIP_MEMORY_SCOPE_AGENT);
}

// ---------------------------------------------------------------------------
// Core WMMA inner product for one 16-col tile: acc += A[16xK] @ W[K,kb0:kb1)
// ---------------------------------------------------------------------------
__device__ __forceinline__ v8f wmma_tile(const _Float16* __restrict__ A,
                                         const float* __restrict__ W,
                                         int N, int K, int kb0, int kb1,
                                         int r, int hs, int tn)
{
    v8f acc = {};
    const _Float16* arow = A + (size_t)r * K;
    const float*    wcol = W + tn + r;
    for (int kb = kb0; kb < kb1; kb += 32) {
        Frag16 af, bf;
        // A frag: lanes<16 hold K = kb+{0..7,16..23}; lanes>=16 hold +8
        af.h8[0] = *(const v8h*)(arow + kb + hs * 8);
        af.h8[1] = *(const v8h*)(arow + kb + 16 + hs * 8);
        // B frag: lane holds column tn+r; coalesced 64B segments across lanes
        const float* w0 = wcol + (size_t)(kb + hs * 8) * N;
        const float* w1 = w0 + (size_t)16 * N;
#pragma unroll
        for (int j = 0; j < 8; ++j) {
            bf.v[j]     = (_Float16)w0[(size_t)j * N];
            bf.v[j + 8] = (_Float16)w1[(size_t)j * N];
        }
        acc = __builtin_amdgcn_wmma_f32_16x16x32_f16(
            false, af.v, false, bf.v, (short)0, acc, false, false);
    }
    return acc;
}

// ---------------------------------------------------------------------------
// Fused Q/K/V GEMM: grid (N/64, 3 matrices, SPLITK). Atomic-accumulate into
// pre-zeroed outputs. N = K = Dd.
// ---------------------------------------------------------------------------
template <int SPLITK>
__global__ __launch_bounds__(128) void qkv_wmma(
    const _Float16* __restrict__ A,
    const float* __restrict__ Wq, const float* __restrict__ Wk,
    const float* __restrict__ Wv,
    float* __restrict__ qf, float* __restrict__ kf, float* __restrict__ vf)
{
    const int lane = threadIdx.x & 31;
    const int wave = threadIdx.x >> 5;
    const int hs = lane >> 4, r = lane & 15;
    const int tn = (blockIdx.x * 4 + wave) << 4;
    const float* W = (blockIdx.y == 0) ? Wq : (blockIdx.y == 1) ? Wk : Wv;
    float* out     = (blockIdx.y == 0) ? qf : (blockIdx.y == 1) ? kf : vf;
    const int kb0 = blockIdx.z * (Dd / SPLITK);

    v8f acc = wmma_tile(A, W, Dd, Dd, kb0, kb0 + Dd / SPLITK, r, hs, tn);
#pragma unroll
    for (int i = 0; i < 8; ++i)
        atomic_add_f32(&out[(size_t)(i + hs * 8) * Dd + tn + r], acc[i]);
}

// ---------------------------------------------------------------------------
// Generic GEMM. MODE 1: residual atomic add -> f32 (x += A@W)
//               MODE 2: bias + gelu -> f16 (SPLITK must be 1)
//               MODE 3: bias (chunk 0 only) + residual atomic add -> f32
// grid (N/64, SPLITK)
// ---------------------------------------------------------------------------
template <int MODE, int SPLITK>
__global__ __launch_bounds__(128) void gemm16_wmma(
    const _Float16* __restrict__ A, const float* __restrict__ W,
    const float* __restrict__ bias, float* __restrict__ outf,
    _Float16* __restrict__ outh, int N, int K)
{
    const int lane = threadIdx.x & 31;
    const int wave = threadIdx.x >> 5;
    const int hs = lane >> 4, r = lane & 15;
    const int tn = (blockIdx.x * 4 + wave) << 4;
    if (tn >= N) return;
    const int kb0 = blockIdx.y * (K / SPLITK);

    v8f acc = wmma_tile(A, W, N, K, kb0, kb0 + K / SPLITK, r, hs, tn);

#pragma unroll
    for (int i = 0; i < 8; ++i) {
        const int row = i + hs * 8;
        const int col = tn + r;
        const float v = acc[i];
        if constexpr (MODE == 1) {
            if constexpr (SPLITK == 1) outf[(size_t)row * N + col] += v;
            else atomic_add_f32(&outf[(size_t)row * N + col], v);
        } else if constexpr (MODE == 2) {
            float t = v + bias[col];
            float g = 0.5f * t *
                (1.0f + tanhf(0.7978845608f * (t + 0.044715f * t * t * t)));
            outh[(size_t)row * N + col] = (_Float16)g;
        } else {
            float add = v + ((blockIdx.y == 0) ? bias[col] : 0.0f);
            if constexpr (SPLITK == 1) outf[(size_t)row * N + col] += add;
            else atomic_add_f32(&outf[(size_t)row * N + col], add);
        }
    }
}

// ---------------------------------------------------------------------------
// LayerNorm (biased var, eps 1e-5) -> f16 row
// ---------------------------------------------------------------------------
__global__ __launch_bounds__(256) void ln_f16(
    const float* __restrict__ x, const float* __restrict__ g,
    const float* __restrict__ b, _Float16* __restrict__ h)
{
    __shared__ float s1[256], s2[256];
    const int row = blockIdx.x, t = threadIdx.x;
    const float* xr = x + (size_t)row * Dd;
    float a0 = 0.f, a1 = 0.f;
    for (int d = t; d < Dd; d += 256) { float v = xr[d]; a0 += v; a1 += v * v; }
    s1[t] = a0; s2[t] = a1; __syncthreads();
    for (int s = 128; s > 0; s >>= 1) {
        if (t < s) { s1[t] += s1[t + s]; s2[t] += s2[t + s]; }
        __syncthreads();
    }
    const float mean = s1[0] * (1.0f / Dd);
    const float var  = s2[0] * (1.0f / Dd) - mean * mean;
    const float inv  = rsqrtf(var + 1e-5f);
    for (int d = t; d < Dd; d += 256)
        h[(size_t)row * Dd + d] = (_Float16)((xr[d] - mean) * inv * g[d] + b[d]);
}

// ---------------------------------------------------------------------------
// Flash-style split attention. One block per (b, h, m-chunk).
// KV prefix (read-once, 1.2 GB) is loaded NON-TEMPORAL so it does not evict
// the weight working set from the 192 MB L2 between graph replays.
// ---------------------------------------------------------------------------
__global__ __launch_bounds__(256) void attn_part(
    const float* __restrict__ qf, const float* __restrict__ knew,
    const float* __restrict__ vnew, const float* __restrict__ kmem,
    const float* __restrict__ vmem, float* __restrict__ pout,
    float* __restrict__ pmax, float* __restrict__ psum)
{
    __shared__ float sc[ACHUNK];
    __shared__ float red[256];
    const int bh = blockIdx.x / ASPLIT;
    const int s  = blockIdx.x % ASPLIT;
    const int b = bh / Hh, h = bh % Hh;
    const int t = threadIdx.x;
    const int lane = t & 31, wave = t >> 5;
    const int sub = lane >> 4, li = lane & 15;
    const int MT = Mm + 1;
    const int ms = s * ACHUNK;
    const int me = (ms + ACHUNK < MT) ? ms + ACHUNK : MT;
    const int nrows = me - ms;

    const v4f qv = *(const v4f*)(qf + (size_t)b * Dd + h * DHd + li * 4);
    const float* kbase = kmem + (size_t)b * Mm * Dd + h * DHd;
    const float* knrow = knew + (size_t)b * Dd + h * DHd;

    for (int m = ms + wave * 2 + sub; m < me; m += 16) {
        v4f kv;
        if (m < Mm)
            kv = __builtin_nontemporal_load(
                (const v4f*)(kbase + (size_t)m * Dd + li * 4));
        else
            kv = *(const v4f*)(knrow + li * 4);
        float p = qv.x * kv.x + qv.y * kv.y + qv.z * kv.z + qv.w * kv.w;
        p += __shfl_xor(p, 1, 32);
        p += __shfl_xor(p, 2, 32);
        p += __shfl_xor(p, 4, 32);
        p += __shfl_xor(p, 8, 32);
        if (li == 0) sc[m - ms] = p * 0.125f;   // 1/sqrt(64)
    }
    __syncthreads();

    float lm = -3.0e38f;
    for (int i = t; i < nrows; i += 256) lm = fmaxf(lm, sc[i]);
    red[t] = lm; __syncthreads();
    for (int w = 128; w > 0; w >>= 1) {
        if (t < w) red[t] = fmaxf(red[t], red[t + w]);
        __syncthreads();
    }
    const float mx = red[0]; __syncthreads();

    float ls = 0.f;
    for (int i = t; i < nrows; i += 256) { float e = __expf(sc[i] - mx); sc[i] = e; ls += e; }
    red[t] = ls; __syncthreads();
    for (int w = 128; w > 0; w >>= 1) {
        if (t < w) red[t] += red[t + w];
        __syncthreads();
    }
    const float sum = red[0]; __syncthreads();

    const int dh = t & 63, ck = t >> 6;
    const float* vbase = vmem + (size_t)b * Mm * Dd + h * DHd;
    const float* vnrow = vnew + (size_t)b * Dd + h * DHd;
    float acc = 0.f;
    for (int i = ck; i < nrows; i += 4) {
        const int m = ms + i;
        float vv;
        if (m < Mm)
            vv = __builtin_nontemporal_load(vbase + (size_t)m * Dd + dh);
        else
            vv = vnrow[dh];
        acc += sc[i] * vv;
    }
    red[ck * 64 + dh] = acc; __syncthreads();
    if (t < 64)
        pout[((size_t)bh * ASPLIT + s) * 64 + t] =
            red[t] + red[64 + t] + red[128 + t] + red[192 + t];
    if (t == 0) { pmax[bh * ASPLIT + s] = mx; psum[bh * ASPLIT + s] = sum; }
}

__global__ __launch_bounds__(64) void attn_comb(
    const float* __restrict__ pout, const float* __restrict__ pmax,
    const float* __restrict__ psum, _Float16* __restrict__ o16)
{
    const int bh = blockIdx.x, t = threadIdx.x;
    const int b = bh / Hh, h = bh % Hh;
    float gmax = -3.0e38f;
#pragma unroll
    for (int s = 0; s < ASPLIT; ++s) gmax = fmaxf(gmax, pmax[bh * ASPLIT + s]);
    float denom = 0.f, acc = 0.f;
#pragma unroll
    for (int s = 0; s < ASPLIT; ++s) {
        const float f = __expf(pmax[bh * ASPLIT + s] - gmax);
        denom += psum[bh * ASPLIT + s] * f;
        acc   += pout[((size_t)bh * ASPLIT + s) * 64 + t] * f;
    }
    o16[(size_t)b * Dd + h * DHd + t] = (_Float16)(acc / denom);
}

__global__ void embed_gather(const int* __restrict__ tok,
                             const float* __restrict__ emb,
                             float* __restrict__ x)
{
    const int i = blockIdx.x * 256 + threadIdx.x;
    const int b = i / Dd, d = i % Dd;
    x[i] = emb[(size_t)tok[b] * Dd + d];
}

__global__ void zero16(uint4* __restrict__ p, int n)
{
    const int i = blockIdx.x * 256 + threadIdx.x;
    if (i < n) p[i] = uint4{0u, 0u, 0u, 0u};
}

__global__ __launch_bounds__(256) void logits_k(
    const float* __restrict__ x, const float* __restrict__ Wout,
    const float* __restrict__ bout, float* __restrict__ out)
{
    const int t = blockIdx.x * 256 + threadIdx.x;
    if (t >= Bq * Vv) return;
    const int b = t / Vv, v = t % Vv;
    float acc = bout[v];
    for (int k = 0; k < Dd; ++k) acc += x[(size_t)b * Dd + k] * Wout[(size_t)k * Vv + v];
    out[t] = acc;
}

// ---------------------------------------------------------------------------
extern "C" void kernel_launch(void* const* d_in, const int* in_sizes, int n_in,
                              void* d_out, int out_size, void* d_ws, size_t ws_size,
                              hipStream_t stream)
{
    const int*   tokens = (const int*)d_in[0];
    const float* embedW = (const float*)d_in[1];
    const float* ln1_g = (const float*)d_in[2];
    const float* ln1_b = (const float*)d_in[3];
    const float* ln2_g = (const float*)d_in[4];
    const float* ln2_b = (const float*)d_in[5];
    const float* ln3_g = (const float*)d_in[6];
    const float* ln3_b = (const float*)d_in[7];
    const float* Wq_s = (const float*)d_in[8];
    const float* Wk_s = (const float*)d_in[9];
    const float* Wv_s = (const float*)d_in[10];
    const float* Wo_s = (const float*)d_in[11];
    const float* Wq_c = (const float*)d_in[12];
    const float* Wk_c = (const float*)d_in[13];
    const float* Wv_c = (const float*)d_in[14];
    const float* Wo_c = (const float*)d_in[15];
    const float* W1   = (const float*)d_in[16];
    const float* b1   = (const float*)d_in[17];
    const float* W2   = (const float*)d_in[18];
    const float* b2   = (const float*)d_in[19];
    const float* skm  = (const float*)d_in[20];
    const float* svm  = (const float*)d_in[21];
    const float* ckm  = (const float*)d_in[22];
    const float* cvm  = (const float*)d_in[23];
    const float* Wout = (const float*)d_in[24];
    const float* bout = (const float*)d_in[25];

    // Workspace layout (16-row padded activations)
    float* x  = (float*)d_ws;                    // 16 x 768 f32
    float* qf = x  + 16 * Dd;                    // 3 x (16 x 768) f32, contiguous
    float* kf = qf + 16 * Dd;
    float* vf = kf + 16 * Dd;
    _Float16* h16 = (_Float16*)(vf + 16 * Dd);   // 16 x 768 f16
    _Float16* o16 = h16 + 16 * Dd;               // 16 x 768 f16
    _Float16* g16 = o16 + 16 * Dd;               // 16 x 3072 f16
    float* pout = (float*)(g16 + 16 * FFd);      // B*H*ASPLIT*64
    float* pmax = pout + Bq * Hh * ASPLIT * 64;
    float* psum = pmax + Bq * Hh * ASPLIT;

    // Zero f16 staging (WMMA pad rows must stay 0)
    const int n16 = (16 * Dd * 2 + 16 * FFd) * 2 / 16;   // 9216 uint4
    zero16<<<(n16 + 255) / 256, 256, 0, stream>>>((uint4*)h16, n16);
    embed_gather<<<(Bq * Dd) / 256, 256, 0, stream>>>(tokens, embedW, x);

    const size_t WSZ = (size_t)Dd * Dd;
    const size_t KVL = (size_t)Bq * Mm * Dd;
    const int nqkv = 3 * 16 * Dd * 4 / 16;               // qf..vf as uint4

    for (int l = 0; l < L_LAYERS; ++l) {
        struct Blk {
            const float *g, *b, *Wq, *Wk, *Wv, *Wo, *km, *vm;
        } blk[2] = {
            { ln1_g + l * Dd, ln1_b + l * Dd,
              Wq_s + l * WSZ, Wk_s + l * WSZ, Wv_s + l * WSZ, Wo_s + l * WSZ,
              skm + l * KVL, svm + l * KVL },
            { ln2_g + l * Dd, ln2_b + l * Dd,
              Wq_c + l * WSZ, Wk_c + l * WSZ, Wv_c + l * WSZ, Wo_c + l * WSZ,
              ckm + l * KVL, cvm + l * KVL },
        };
        for (int a = 0; a < 2; ++a) {
            ln_f16<<<Bq, 256, 0, stream>>>(x, blk[a].g, blk[a].b, h16);
            zero16<<<(nqkv + 255) / 256, 256, 0, stream>>>((uint4*)qf, nqkv);
            qkv_wmma<2><<<dim3(Dd / 64, 3, 2), 128, 0, stream>>>(
                h16, blk[a].Wq, blk[a].Wk, blk[a].Wv, qf, kf, vf);
            attn_part<<<Bq * Hh * ASPLIT, 256, 0, stream>>>(
                qf, kf, vf, blk[a].km, blk[a].vm, pout, pmax, psum);
            attn_comb<<<Bq * Hh, 64, 0, stream>>>(pout, pmax, psum, o16);
            gemm16_wmma<1, 4><<<dim3(Dd / 64, 4), 128, 0, stream>>>(
                o16, blk[a].Wo, nullptr, x, nullptr, Dd, Dd);
        }
        ln_f16<<<Bq, 256, 0, stream>>>(x, ln3_g + l * Dd, ln3_b + l * Dd, h16);
        gemm16_wmma<2, 1><<<dim3(FFd / 64, 1), 128, 0, stream>>>(
            h16, W1 + (size_t)l * Dd * FFd, b1 + (size_t)l * FFd,
            nullptr, g16, FFd, Dd);
        gemm16_wmma<3, 4><<<dim3(Dd / 64, 4), 128, 0, stream>>>(
            g16, W2 + (size_t)l * FFd * Dd, b2 + (size_t)l * Dd,
            x, nullptr, Dd, FFd);
    }
    logits_k<<<1, 256, 0, stream>>>(x, Wout, bout, (float*)d_out);
}
